// AxialAttention_5669356834377
// MI455X (gfx1250) — matrix operand order repaired and checked by reading, not compiled, not run on hardware
//
#include <hip/hip_runtime.h>

// ---------------- types for WMMA fragments ----------------
typedef __attribute__((ext_vector_type(16))) __bf16 v16bf;
typedef __attribute__((ext_vector_type(8)))  float  v8f;
typedef __attribute__((ext_vector_type(4)))  int    v4i;

union Frag16 {
    v4i   q[2];   // 2 x 16B = 32B = 16 bf16
    v16bf v;
};

__device__ __forceinline__ unsigned short f2bf(float f) {
    unsigned u = __float_as_uint(f);
    unsigned r = u + 0x7FFFu + ((u >> 16) & 1u);   // round-to-nearest-even
    return (unsigned short)(r >> 16);
}

// ---------------- dims ----------------
#define NN 16
#define CIN 512
#define HH 56
#define SS 3136          // 56*56
#define OC 1024          // 2*OUT_PLANES
#define NGROUP 8
#define NB 896           // N*W
#define NBG 7168         // NB*NGROUP

// ============ 1) convert w_qkv -> bf16 (row-major o x c, already A layout) ============
__global__ void k_cvt_bf16(const float* __restrict__ src, unsigned short* __restrict__ dst, int n) {
    for (int i = blockIdx.x * blockDim.x + threadIdx.x; i < n; i += gridDim.x * blockDim.x)
        dst[i] = f2bf(src[i]);
}

// ============ 2) transpose x[n] (512 x 3136) -> xt16[n] (3136 x 512) bf16 ============
__global__ void k_xpose(const float* __restrict__ x, unsigned short* __restrict__ xt) {
    __shared__ float tile[32][33];
    int n  = blockIdx.z;
    int c0 = blockIdx.y * 32, s0 = blockIdx.x * 32;
    const float* xp = x + (size_t)n * CIN * SS;
    unsigned short* xtp = xt + (size_t)n * SS * CIN;
    int tx = threadIdx.x, ty = threadIdx.y;      // (32, 8)
    for (int r = ty; r < 32; r += 8)
        tile[r][tx] = xp[(size_t)(c0 + r) * SS + s0 + tx];
    __syncthreads();
    for (int r = ty; r < 32; r += 8)
        xtp[(size_t)(s0 + r) * CIN + c0 + tx] = f2bf(tile[tx][r]);
}

// ============ 3) qkv GEMM: qkv[n][o][s] = sum_c w[o][c] * x[n][c][s]  (WMMA bf16) ============
// block = 128 threads (4 waves); wave computes 32x32; block covers 64(M) x 64(N)
__global__ void k_gemm_qkv(const unsigned short* __restrict__ w16,
                           const unsigned short* __restrict__ xt16,
                           float* __restrict__ qkv) {
    int n     = blockIdx.z;
    int m_blk = blockIdx.y * 64;   // o
    int s_blk = blockIdx.x * 64;   // s
    int wave  = threadIdx.x >> 5;
    int lane  = threadIdx.x & 31;
    int wm = (wave >> 1) * 32;
    int wn = (wave & 1) * 32;
    int m = lane & 15, half = lane >> 4;

    const unsigned short* A = w16;                              // 1024 x 512, K contiguous
    const unsigned short* B = xt16 + (size_t)n * SS * CIN;      // 3136 x 512, K contiguous (acts as B^T)

    int am0 = m_blk + wm + m;        // A rows for mt=0/1
    int am1 = am0 + 16;
    int bn0 = s_blk + wn + m;        // B^T rows for nt=0/1
    int bn1 = bn0 + 16;

    v8f acc00 = {}, acc01 = {}, acc10 = {}, acc11 = {};

    for (int k0 = 0; k0 < CIN; k0 += 32) {
        Frag16 a0, a1, b0, b1;
        // A fragments: two 16B chunks at K = k0 + half*8 and k0 + 16 + half*8
        {
            const unsigned short* p = A + (size_t)am0 * CIN + k0 + half * 8;
            a0.q[0] = *(const v4i*)p; a0.q[1] = *(const v4i*)(p + 16);
        }
        {
            const unsigned short* p = A + (size_t)am1 * CIN + k0 + half * 8;
            a1.q[0] = *(const v4i*)p; a1.q[1] = *(const v4i*)(p + 16);
        }
        // B fragments: two 16B chunks at K = k0 + half*8 and k0 + 16 + half*8
        {
            const unsigned short* p = B + (size_t)bn0 * CIN + k0 + half * 8;
            b0.q[0] = *(const v4i*)p; b0.q[1] = *(const v4i*)(p + 16);
        }
        {
            const unsigned short* p = B + (size_t)bn1 * CIN + k0 + half * 8;
            b1.q[0] = *(const v4i*)p; b1.q[1] = *(const v4i*)(p + 16);
        }
        acc00 = __builtin_amdgcn_wmma_f32_16x16x32_bf16(false, a0.v, false, b0.v, (short)0, acc00, false, false);
        acc01 = __builtin_amdgcn_wmma_f32_16x16x32_bf16(false, a0.v, false, b1.v, (short)0, acc01, false, false);
        acc10 = __builtin_amdgcn_wmma_f32_16x16x32_bf16(false, a1.v, false, b0.v, (short)0, acc10, false, false);
        acc11 = __builtin_amdgcn_wmma_f32_16x16x32_bf16(false, a1.v, false, b1.v, (short)0, acc11, false, false);
    }

    float* qn = qkv + (size_t)n * OC * SS;
    int sc0 = s_blk + wn + m;
#pragma unroll
    for (int r = 0; r < 8; ++r) {
        int mm = m_blk + wm + r + 8 * half;
        qn[(size_t)mm * SS + sc0]             = acc00[r];
        qn[(size_t)mm * SS + sc0 + 16]        = acc01[r];
        qn[(size_t)(mm + 16) * SS + sc0]      = acc10[r];
        qn[(size_t)(mm + 16) * SS + sc0 + 16] = acc11[r];
    }
}

// ============ 4) generic per-channel BN stats -> affine scale/shift ============
__global__ void k_bn_stats(const float* __restrict__ buf, size_t ch_stride, size_t outer_stride,
                           int outer_cnt, int inner_cnt,
                           const float* __restrict__ gamma, const float* __restrict__ beta,
                           float* __restrict__ scale, float* __restrict__ shift) {
    int c = blockIdx.x;
    const float* p = buf + (size_t)c * ch_stride;
    float s = 0.f, s2 = 0.f;
    for (int o = 0; o < outer_cnt; ++o) {
        const float* q = p + (size_t)o * outer_stride;
        for (int i = threadIdx.x; i < inner_cnt; i += blockDim.x) {
            float v = q[i]; s += v; s2 += v * v;
        }
    }
    __shared__ float rs[256], rs2[256];
    rs[threadIdx.x] = s; rs2[threadIdx.x] = s2;
    __syncthreads();
    for (int d = blockDim.x >> 1; d > 0; d >>= 1) {
        if ((int)threadIdx.x < d) { rs[threadIdx.x] += rs[threadIdx.x + d]; rs2[threadIdx.x] += rs2[threadIdx.x + d]; }
        __syncthreads();
    }
    if (threadIdx.x == 0) {
        float cnt  = (float)outer_cnt * (float)inner_cnt;
        float mean = rs[0] / cnt;
        float var  = rs2[0] / cnt - mean * mean;
        float sc   = gamma[c] * rsqrtf(var + 1e-5f);
        scale[c] = sc;
        shift[c] = beta[c] - mean * sc;
    }
}

// ============ 5) build embeddings: emb[c][i][j] = relative[c][i-j+55] ============
__global__ void k_emb(const float* __restrict__ relative, float* __restrict__ emb) {
    int idx = blockIdx.x * blockDim.x + threadIdx.x;
    if (idx >= 128 * SS) return;
    int c = idx / SS, ij = idx % SS;
    int i = ij / HH, j = ij % HH;
    emb[idx] = relative[c * 111 + (i - j + 55)];
}

// ============ 6) qk[b][g][i][j] = sum_c q[c,i]*k[c,j]  (WMMA, LDS-staged) ============
__global__ void k_qk(const float* __restrict__ qkv, const float* __restrict__ scq,
                     const float* __restrict__ shq, float* __restrict__ qkbuf) {
    int bg = blockIdx.x;
    int g = bg & 7, b = bg >> 3;
    int n = b / HH, w = b % HH;
    __shared__ unsigned short qT[64 * 32];   // rows i, cols c (K-contiguous)
    __shared__ unsigned short kT[64 * 32];   // rows j, cols c
    int tid = threadIdx.x;
    for (int idx = tid; idx < 64 * 32; idx += 128) { qT[idx] = 0; kT[idx] = 0; }
    __syncthreads();
    const float* base = qkv + ((size_t)n * OC + g * 128) * SS + w;
    for (int idx = tid; idx < 32 * HH; idx += 128) {
        int c = idx / HH, i = idx % HH;
        float qv = base[(size_t)c * SS + i * HH]        * scq[g * 128 + c]      + shq[g * 128 + c];
        float kv = base[(size_t)(32 + c) * SS + i * HH] * scq[g * 128 + 32 + c] + shq[g * 128 + 32 + c];
        qT[i * 32 + c] = f2bf(qv);
        kT[i * 32 + c] = f2bf(kv);
    }
    __syncthreads();
    int wave = tid >> 5, lane = tid & 31;
    int m = lane & 15, half = lane >> 4;
    int i0 = wave * 16;
    Frag16 a;
    {
        const unsigned short* p = &qT[(i0 + m) * 32 + half * 8];
        a.q[0] = *(const v4i*)p; a.q[1] = *(const v4i*)(p + 16);
    }
    float* outb = qkbuf + (size_t)bg * SS;
#pragma unroll
    for (int jt = 0; jt < 4; ++jt) {
        Frag16 bb;
        const unsigned short* p = &kT[(jt * 16 + m) * 32 + half * 8];
        bb.q[0] = *(const v4i*)p; bb.q[1] = *(const v4i*)(p + 16);
        v8f acc = {};
        acc = __builtin_amdgcn_wmma_f32_16x16x32_bf16(false, a.v, false, bb.v, (short)0, acc, false, false);
#pragma unroll
        for (int r = 0; r < 8; ++r) {
            int ii = i0 + r + 8 * half;
            int jj = jt * 16 + m;
            if (ii < HH && jj < HH) outb[ii * HH + jj] = acc[r];
        }
    }
}

// ============ 7) qr / kr (i-dependent einsums, VALU) ============
__global__ void k_qrkr(const float* __restrict__ qkv, const float* __restrict__ scq,
                       const float* __restrict__ shq, const float* __restrict__ emb,
                       float* __restrict__ qrbuf, float* __restrict__ krbuf) {
    size_t total = (size_t)NBG * SS;
    for (size_t idx = (size_t)blockIdx.x * blockDim.x + threadIdx.x; idx < total;
         idx += (size_t)gridDim.x * blockDim.x) {
        int ij = (int)(idx % SS);
        int bg = (int)(idx / SS);
        int g = bg & 7, b = bg >> 3;
        int n = b / HH, w = b % HH;
        int i = ij / HH, j = ij % HH;
        const float* base = qkv + ((size_t)n * OC + g * 128) * SS + w;
        float accq = 0.f, acck = 0.f;
        for (int c = 0; c < 32; ++c) {
            float qv = base[(size_t)c * SS + i * HH] * scq[g * 128 + c] + shq[g * 128 + c];
            accq += qv * emb[(size_t)c * SS + ij];                           // q_emb[c][i][j]
            float kv = base[(size_t)(32 + c) * SS + j * HH] * scq[g * 128 + 32 + c] + shq[g * 128 + 32 + c];
            acck += kv * emb[(size_t)(32 + c) * SS + j * HH + i];            // k_emb[c][j][i]
        }
        qrbuf[idx] = accq;
        krbuf[idx] = acck;
    }
}

// ============ 8) BN(stacked) + sum + softmax over j ============
__global__ void k_softmax(const float* __restrict__ qk, const float* __restrict__ qr,
                          const float* __restrict__ kr, const float* __restrict__ scs,
                          const float* __restrict__ shs, float* __restrict__ sim) {
    int row = blockIdx.x;                 // bg*56 + i
    int i = row % HH;
    int bg = row / HH;
    int g = bg & 7;
    size_t off = (size_t)bg * SS + i * HH;
    __shared__ float red[64];
    int t = threadIdx.x;
    bool act = (t < HH);
    float sv = 0.f;
    if (act) {
        sv = qk[off + t] * scs[g]      + shs[g]
           + qr[off + t] * scs[8 + g]  + shs[8 + g]
           + kr[off + t] * scs[16 + g] + shs[16 + g];
    }
    red[t] = act ? sv : -3.4e38f;
    __syncthreads();
    for (int d = 32; d > 0; d >>= 1) {
        if (t < d) red[t] = fmaxf(red[t], red[t + d]);
        __syncthreads();
    }
    float mx = red[0];
    __syncthreads();
    float e = act ? __expf(sv - mx) : 0.f;
    red[t] = e;
    __syncthreads();
    for (int d = 32; d > 0; d >>= 1) {
        if (t < d) red[t] += red[t + d];
        __syncthreads();
    }
    float inv = 1.f / red[0];
    if (act) sim[off + t] = e * inv;
}

// ============ 9) sv (WMMA) + sve (VALU), write out_pre[b][g*128+2c+s][i] ============
__global__ void k_sv(const float* __restrict__ qkv, const float* __restrict__ scq,
                     const float* __restrict__ shq, const float* __restrict__ sim,
                     const float* __restrict__ emb, float* __restrict__ outpre) {
    int bg = blockIdx.x;
    int g = bg & 7, b = bg >> 3;
    int n = b / HH, w = b % HH;
    __shared__ unsigned short vS[64 * 64];   // rows c, cols j (K-contiguous)
    __shared__ unsigned short sS[64 * 64];   // rows i, cols j (acts as B^T)
    int tid = threadIdx.x;
    for (int idx = tid; idx < 64 * 64; idx += 256) { vS[idx] = 0; sS[idx] = 0; }
    __syncthreads();
    const float* vb = qkv + ((size_t)n * OC + g * 128 + 64) * SS + w;
    for (int idx = tid; idx < 64 * HH; idx += 256) {
        int c = idx / HH, j = idx % HH;
        float vv = vb[(size_t)c * SS + j * HH] * scq[g * 128 + 64 + c] + shq[g * 128 + 64 + c];
        vS[c * 64 + j] = f2bf(vv);
    }
    const float* sb = sim + (size_t)bg * SS;
    for (int idx = tid; idx < SS; idx += 256) {
        int i = idx / HH, j = idx % HH;
        sS[i * 64 + j] = f2bf(sb[idx]);
    }
    __syncthreads();
    int wave = tid >> 5, lane = tid & 31;
    int m = lane & 15, half = lane >> 4;
#pragma unroll
    for (int t = 0; t < 2; ++t) {
        int tt = wave * 2 + t;
        int ct = tt >> 2, it = tt & 3;
        v8f acc = {};
#pragma unroll
        for (int k0 = 0; k0 < 64; k0 += 32) {
            Frag16 a, bb;
            const unsigned short* pa = &vS[(ct * 16 + m) * 64 + k0 + half * 8];
            a.q[0] = *(const v4i*)pa; a.q[1] = *(const v4i*)(pa + 16);
            const unsigned short* pb = &sS[(it * 16 + m) * 64 + k0 + half * 8];
            bb.q[0] = *(const v4i*)pb; bb.q[1] = *(const v4i*)(pb + 16);
            acc = __builtin_amdgcn_wmma_f32_16x16x32_bf16(false, a.v, false, bb.v, (short)0, acc, false, false);
        }
#pragma unroll
        for (int r = 0; r < 8; ++r) {
            int c = ct * 16 + r + 8 * half;
            int i = it * 16 + m;
            if (i < HH) outpre[((size_t)b * OC + g * 128 + 2 * c) * HH + i] = acc[r];
        }
    }
    // sve[c][i] = sum_j sim[i][j] * v_emb[c][i][j]
    for (int idx = tid; idx < 64 * HH; idx += 256) {
        int c = idx / HH, i = idx % HH;
        const float* ve = emb + (size_t)(64 + c) * SS + i * HH;
        const float* sr = sb + i * HH;
        float acc2 = 0.f;
        for (int j = 0; j < HH; ++j) acc2 += sr[j] * ve[j];
        outpre[((size_t)b * OC + g * 128 + 2 * c + 1) * HH + i] = acc2;
    }
}

// ============ 10) final BN + pair-sum + transpose to (N, 512, H, W) ============
__global__ void k_final(const float* __restrict__ outpre, const float* __restrict__ sco,
                        const float* __restrict__ sho, float* __restrict__ out) {
    size_t total = (size_t)NN * 512 * SS;
    for (size_t idx = (size_t)blockIdx.x * blockDim.x + threadIdx.x; idx < total;
         idx += (size_t)gridDim.x * blockDim.x) {
        int w = (int)(idx % HH);
        size_t r = idx / HH;
        int h = (int)(r % HH); r /= HH;
        int p = (int)(r % 512);
        int n = (int)(r / 512);
        int b = n * HH + w;
        const float* pp = outpre + ((size_t)b * OC + 2 * p) * HH + h;
        out[idx] = pp[0] * sco[2 * p] + sho[2 * p] + pp[HH] * sco[2 * p + 1] + sho[2 * p + 1];
    }
}

extern "C" void kernel_launch(void* const* d_in, const int* in_sizes, int n_in,
                              void* d_out, int out_size, void* d_ws, size_t ws_size,
                              hipStream_t stream) {
    (void)in_sizes; (void)n_in; (void)out_size; (void)ws_size;
    const float* x        = (const float*)d_in[0];
    const float* w_qkv    = (const float*)d_in[1];
    const float* relative = (const float*)d_in[2];
    const float* g_qkv    = (const float*)d_in[3];
    const float* b_qkv    = (const float*)d_in[4];
    const float* g_sim    = (const float*)d_in[5];
    const float* b_sim    = (const float*)d_in[6];
    const float* g_out    = (const float*)d_in[7];
    const float* b_out    = (const float*)d_in[8];
    float* out = (float*)d_out;

    char* ws = (char*)d_ws;
    size_t off = 0;
    auto alloc = [&](size_t bytes) -> void* {
        void* p = ws + off;
        off += (bytes + 255) & ~(size_t)255;
        return p;
    };
    unsigned short* w16  = (unsigned short*)alloc((size_t)OC * CIN * 2);
    unsigned short* xt16 = (unsigned short*)alloc((size_t)NN * SS * CIN * 2);
    float* qkv    = (float*)alloc((size_t)NN * OC * SS * 4);
    float* sc_qkv = (float*)alloc(OC * 4);
    float* sh_qkv = (float*)alloc(OC * 4);
    float* emb    = (float*)alloc((size_t)128 * SS * 4);
    float* qkb    = (float*)alloc((size_t)NBG * SS * 4);
    float* qrb    = (float*)alloc((size_t)NBG * SS * 4);
    float* krb    = (float*)alloc((size_t)NBG * SS * 4);
    float* sc_sim = (float*)alloc(24 * 4);
    float* sh_sim = (float*)alloc(24 * 4);
    float* simb   = (float*)alloc((size_t)NBG * SS * 4);
    float* outpre = (float*)alloc((size_t)NB * OC * HH * 4);
    float* sc_out = (float*)alloc(OC * 4);
    float* sh_out = (float*)alloc(OC * 4);

    k_cvt_bf16<<<512, 256, 0, stream>>>(w_qkv, w16, OC * CIN);
    {
        dim3 g(SS / 32, CIN / 32, NN), tb(32, 8);
        k_xpose<<<g, tb, 0, stream>>>(x, xt16);
    }
    {
        dim3 g(SS / 64, OC / 64, NN);
        k_gemm_qkv<<<g, 128, 0, stream>>>(w16, xt16, qkv);
    }
    k_bn_stats<<<OC, 256, 0, stream>>>(qkv, SS, (size_t)OC * SS, NN, SS, g_qkv, b_qkv, sc_qkv, sh_qkv);
    k_emb<<<(128 * SS + 255) / 256, 256, 0, stream>>>(relative, emb);
    k_qk<<<NBG, 128, 0, stream>>>(qkv, sc_qkv, sh_qkv, qkb);
    k_qrkr<<<8192, 256, 0, stream>>>(qkv, sc_qkv, sh_qkv, emb, qrb, krb);
    k_bn_stats<<<NGROUP, 256, 0, stream>>>(qkb, SS, (size_t)NGROUP * SS, NB, SS, g_sim,      b_sim,      sc_sim,      sh_sim);
    k_bn_stats<<<NGROUP, 256, 0, stream>>>(qrb, SS, (size_t)NGROUP * SS, NB, SS, g_sim + 8,  b_sim + 8,  sc_sim + 8,  sh_sim + 8);
    k_bn_stats<<<NGROUP, 256, 0, stream>>>(krb, SS, (size_t)NGROUP * SS, NB, SS, g_sim + 16, b_sim + 16, sc_sim + 16, sh_sim + 16);
    k_softmax<<<NBG * HH, 64, 0, stream>>>(qkb, qrb, krb, sc_sim, sh_sim, simb);
    k_sv<<<NBG, 256, 0, stream>>>(qkv, sc_qkv, sh_qkv, simb, emb, outpre);
    k_bn_stats<<<OC, 256, 0, stream>>>(outpre, HH, (size_t)OC * HH, NB, HH, g_out, b_out, sc_out, sh_out);
    k_final<<<8192, 256, 0, stream>>>(outpre, sc_out, sh_out, out);
}
